// CrossAttentionBlock_78881369358741
// MI455X (gfx1250) — matrix operand order repaired and checked by reading, not compiled, not run on hardware
//
#include <hip/hip_runtime.h>
#include <hip/hip_bf16.h>

typedef _Float16 v16h __attribute__((ext_vector_type(16)));
typedef _Float16 v8h  __attribute__((ext_vector_type(8)));
typedef _Float16 v4h  __attribute__((ext_vector_type(4)));
typedef float    v8f  __attribute__((ext_vector_type(8)));

constexpr int kC   = 768;     // channels
constexpr int kB   = 16;      // batch
constexpr int kN   = 4096;    // img tokens
constexpr int kH   = 12;      // heads
constexpr int kHP  = 16;      // heads padded to 16 for WMMA N
constexpr int kP   = 128;
constexpr int kO   = 64;
constexpr float kEPS   = 1e-5f;
constexpr float kSCALE = 0.125f;  // 1/sqrt(64)

// ---------------------------------------------------------------------------
// Prep kernel: per batch, compute
//   Gt[b][h][i]  (f16, h<16):  h<12 -> scale * sum_d wq[i][h*64+d]*(k0-k1)[h*64+d]
//   B2t[b][j][k] (f16, k<32):  k<12 -> sum_d (v0-v1)[k*64+d]*w_out[k*64+d][j]
//                              k==12 -> b_out[j] + sum_i v1[i]*w_out[i][j]
//                              else  -> 0
// ---------------------------------------------------------------------------
__global__ __launch_bounds__(256) void ca_prep(
    const float* __restrict__ param, const float* __restrict__ obj,
    const float* __restrict__ cnw,   const float* __restrict__ cnb,
    const float* __restrict__ wq,    const float* __restrict__ w_param,
    const float* __restrict__ b_param, const float* __restrict__ w_obj,
    const float* __restrict__ b_obj, const float* __restrict__ w_kv,
    const float* __restrict__ w_out, const float* __restrict__ b_out,
    _Float16* __restrict__ wsG, _Float16* __restrict__ wsB)
{
  __shared__ float ctx[2][kC];
  __shared__ float lnc[2][kC];
  __shared__ float kb[2][kC];
  __shared__ float vb[2][kC];
  __shared__ float dk[kC];
  __shared__ float dv[kC];
  __shared__ float red[256];

  const int b = blockIdx.x;
  const int t = threadIdx.x;

  // context projections: param/obj -> [2, C]
  for (int j = t; j < kC; j += 256) {
    float s0 = b_param[j];
    for (int i = 0; i < kP; ++i) s0 += param[b*kP + i] * w_param[i*kC + j];
    ctx[0][j] = s0;
    float s1 = b_obj[j];
    for (int i = 0; i < kO; ++i) s1 += obj[b*kO + i] * w_obj[i*kC + j];
    ctx[1][j] = s1;
  }
  __syncthreads();

  // layernorm on the two context rows
  for (int tt = 0; tt < 2; ++tt) {
    float s = 0.f;
    for (int j = t; j < kC; j += 256) s += ctx[tt][j];
    red[t] = s; __syncthreads();
    for (int off = 128; off > 0; off >>= 1) { if (t < off) red[t] += red[t+off]; __syncthreads(); }
    float mean = red[0] * (1.0f / kC);
    __syncthreads();
    s = 0.f;
    for (int j = t; j < kC; j += 256) { float d = ctx[tt][j] - mean; s += d * d; }
    red[t] = s; __syncthreads();
    for (int off = 128; off > 0; off >>= 1) { if (t < off) red[t] += red[t+off]; __syncthreads(); }
    float rstd = rsqrtf(red[0] * (1.0f / kC) + kEPS);
    __syncthreads();
    for (int j = t; j < kC; j += 256)
      lnc[tt][j] = (ctx[tt][j] - mean) * rstd * cnw[j] + cnb[j];
  }
  __syncthreads();

  // kv = lnc @ w_kv   -> k,v [2, C]
  for (int j = t; j < kC; j += 256) {
    for (int tt = 0; tt < 2; ++tt) {
      float kk = 0.f, vv = 0.f;
      for (int i = 0; i < kC; ++i) {
        float x = lnc[tt][i];
        kk += x * w_kv[i*2*kC + j];
        vv += x * w_kv[i*2*kC + kC + j];
      }
      kb[tt][j] = kk; vb[tt][j] = vv;
    }
  }
  __syncthreads();
  for (int j = t; j < kC; j += 256) {
    dk[j] = kb[0][j] - kb[1][j];
    dv[j] = vb[0][j] - vb[1][j];
  }
  __syncthreads();

  // Gt[b][h][i]
  for (int idx = t; idx < kHP*kC; idx += 256) {
    int h = idx / kC, i = idx - h*kC;
    float val = 0.f;
    if (h < kH) {
      for (int d = 0; d < 64; ++d) val += wq[i*kC + h*64 + d] * dk[h*64 + d];
      val *= kSCALE;
    }
    wsG[(size_t)b*kHP*kC + idx] = (_Float16)val;
  }
  // B2t[b][j][k]
  for (int idx = t; idx < kC*32; idx += 256) {
    int j = idx >> 5, k = idx & 31;
    float val = 0.f;
    if (k < kH) {
      for (int d = 0; d < 64; ++d) val += dv[k*64 + d] * w_out[(k*64 + d)*kC + j];
    } else if (k == kH) {  // k == 12: base row
      val = b_out[j];
      for (int i = 0; i < kC; ++i) val += vb[1][i] * w_out[i*kC + j];
    }
    wsB[(size_t)b*kC*32 + idx] = (_Float16)val;
  }
}

// ---------------------------------------------------------------------------
// Main kernel: per 16-row tile (one wave each):
//   LN(x) -> f16 A-tile in LDS
//   scores[16x16] = A @ Gt^T   (24x v_wmma_f32_16x16x32_f16, K=768)
//   a0 = sigmoid(scores); A2[16x32] = [a0_0..a0_11, 1, 0...]
//   out[16x768] = A2 @ B2      (48x v_wmma_f32_16x16x32_f16), + x residual
// ---------------------------------------------------------------------------
__global__ __launch_bounds__(64) void ca_main(
    const float* __restrict__ img,
    const float* __restrict__ nw, const float* __restrict__ nb,
    const _Float16* __restrict__ wsG, const _Float16* __restrict__ wsB,
    float* __restrict__ out)
{
  __shared__ _Float16 aTile[2][16][kC];     // 48 KB
  __shared__ float    sTile[2][16][16];     //  2 KB

  const int b    = blockIdx.x >> 7;         // 128 blocks per batch
  const int tile = blockIdx.x & 127;        // 32 rows per block (2 waves x 16)
  const int w    = threadIdx.x >> 5;        // wave in block
  const int l    = threadIdx.x & 31;        // lane
  const int r    = l & 15;                  // row within tile / matrix row m
  const int hh   = l >> 4;                  // which 384-column half this lane owns
  const int rowBase = tile*32 + w*16;

  const size_t rowG = (size_t)(b*kN + rowBase + r);
  const float* xr = img + rowG*kC + hh*384;

  // -- layernorm stats: each lane sums 384 columns, pair (l, l^16) covers row r
  float sum = 0.f, sq = 0.f;
  const float4* x4 = reinterpret_cast<const float4*>(xr);
  #pragma unroll 4
  for (int c = 0; c < 96; ++c) {
    float4 v = x4[c];
    sum += v.x + v.y + v.z + v.w;
    sq  += v.x*v.x + v.y*v.y + v.z*v.z + v.w*v.w;
  }
  sum += __shfl_xor(sum, 16, 32);
  sq  += __shfl_xor(sq,  16, 32);
  float mean = sum * (1.0f / kC);
  float rstd = rsqrtf(sq * (1.0f / kC) - mean*mean + kEPS);

  // -- normalize, scale/shift, convert to f16 A-tile in LDS
  const float4* w4 = reinterpret_cast<const float4*>(nw + hh*384);
  const float4* b4 = reinterpret_cast<const float4*>(nb + hh*384);
  #pragma unroll 4
  for (int c = 0; c < 96; ++c) {
    float4 v = x4[c], g = w4[c], bb = b4[c];
    v4h hv;
    hv[0] = (_Float16)((v.x - mean) * rstd * g.x + bb.x);
    hv[1] = (_Float16)((v.y - mean) * rstd * g.y + bb.y);
    hv[2] = (_Float16)((v.z - mean) * rstd * g.z + bb.z);
    hv[3] = (_Float16)((v.w - mean) * rstd * g.w + bb.w);
    *reinterpret_cast<v4h*>(&aTile[w][r][hh*384 + c*4]) = hv;
  }
  __syncthreads();

  const int loffA = (l >= 16) ? 8  : 0;     // A-layout K offset for upper lanes
  const int loffB = (l >= 16) ? 8  : 0;     // B-layout K offset for upper lanes

  // -- stage 1: score tile = LN(x)[16x768] @ G[768x16]
  const _Float16* gB = wsG + (size_t)b*kHP*kC + (size_t)(l & 15)*kC + loffB;
  v8f acc = {};
  for (int ch = 0; ch < 24; ++ch) {
    // A fragment: lane holds row m=l&15, halves 0..7 -> k0..k0+7, 8..15 -> k0+16..k0+23
    v8h lo = *reinterpret_cast<const v8h*>(&aTile[w][r][ch*32 + loffA]);
    v8h hi = *reinterpret_cast<const v8h*>(&aTile[w][r][ch*32 + loffA + 16]);
    v16h a;
    #pragma unroll
    for (int i = 0; i < 8; ++i) { a[i] = lo[i]; a[i+8] = hi[i]; }
    // B fragment: lane holds column n=l&15, halves 0..7 -> k0..k0+7, 8..15 -> k0+16..k0+23
    v8h blo = *reinterpret_cast<const v8h*>(gB + ch*32);
    v8h bhi = *reinterpret_cast<const v8h*>(gB + ch*32 + 16);
    v16h bm;
    #pragma unroll
    for (int i = 0; i < 8; ++i) { bm[i] = blo[i]; bm[i+8] = bhi[i]; }
    acc = __builtin_amdgcn_wmma_f32_16x16x32_f16(false, a, false, bm,
                                                 (short)0, acc, false, false);
  }

  // -- softmax over 2 contexts == sigmoid(s0 - s1); scatter to LDS for transpose
  #pragma unroll
  for (int j = 0; j < 8; ++j) {
    float d = acc[j];                       // D layout: m = j + loffA, n = l&15
    sTile[w][j + loffA][l & 15] = 1.0f / (1.0f + __expf(-d));
  }
  __syncthreads();

  // -- build A2 fragment [16 x 32]: k<12 -> a0_k, k==12 -> 1 (base row), else 0
  v16h a2;
  #pragma unroll
  for (int i = 0; i < 16; ++i) {
    int k = ((i < 8) ? i : (8 + i)) + loffA;
    float v = (k < kH) ? sTile[w][l & 15][k] : ((k == kH) ? 1.0f : 0.0f);
    a2[i] = (_Float16)v;
  }

  // -- stage 2: out[16x768] = A2 @ B2, fused residual + store
  const _Float16* bB = wsB + (size_t)b*kC*32 + (size_t)(l & 15)*32 + loffB;
  const size_t outRowBase = (size_t)(b*kN + rowBase);
  for (int nc = 0; nc < 48; ++nc) {
    v8h blo = *reinterpret_cast<const v8h*>(bB + (size_t)nc*16*32);
    v8h bhi = *reinterpret_cast<const v8h*>(bB + (size_t)nc*16*32 + 16);
    v16h bm;
    #pragma unroll
    for (int i = 0; i < 8; ++i) { bm[i] = blo[i]; bm[i+8] = bhi[i]; }
    v8f z = {};
    v8f o = __builtin_amdgcn_wmma_f32_16x16x32_f16(false, a2, false, bm,
                                                   (short)0, z, false, false);
    int col = nc*16 + (l & 15);
    #pragma unroll
    for (int j = 0; j < 8; ++j) {
      size_t idx = (outRowBase + (size_t)(j + loffA)) * kC + col;
      out[idx] = o[j] + img[idx];
    }
  }
}

// ---------------------------------------------------------------------------
extern "C" void kernel_launch(void* const* d_in, const int* in_sizes, int n_in,
                              void* d_out, int out_size, void* d_ws, size_t ws_size,
                              hipStream_t stream) {
  const float* img      = (const float*)d_in[0];
  const float* param    = (const float*)d_in[1];
  const float* obj      = (const float*)d_in[2];
  const float* inw      = (const float*)d_in[3];
  const float* inb      = (const float*)d_in[4];
  const float* cnw      = (const float*)d_in[5];
  const float* cnb      = (const float*)d_in[6];
  const float* wq       = (const float*)d_in[7];
  const float* w_param  = (const float*)d_in[8];
  const float* b_param  = (const float*)d_in[9];
  const float* w_obj    = (const float*)d_in[10];
  const float* b_obj    = (const float*)d_in[11];
  const float* w_kv     = (const float*)d_in[12];
  const float* w_out    = (const float*)d_in[13];
  const float* b_out    = (const float*)d_in[14];
  float* out = (float*)d_out;

  // workspace: Gt [B][16][768] f16 (384 KB) then B2t [B][768][32] f16 (768 KB)
  _Float16* wsG = (_Float16*)d_ws;
  _Float16* wsB = (_Float16*)((char*)d_ws + (size_t)kB*kHP*kC*sizeof(_Float16));

  ca_prep<<<kB, 256, 0, stream>>>(param, obj, cnw, cnb, wq, w_param, b_param,
                                  w_obj, b_obj, w_kv, w_out, b_out, wsG, wsB);
  ca_main<<<kB * (kN/32), 64, 0, stream>>>(img, inw, inb, wsG, wsB, out);
}
